// AutoLstm_6313601925691
// MI455X (gfx1250) — compile-verified, matrix operand order checked by reading, not run on hardware
//
#include <hip/hip_runtime.h>
#include <stdint.h>

// ---------------------------------------------------------------------------
// Problem constants (V=32000, E=H=512, L=2, B=8, T=1024)
// ---------------------------------------------------------------------------
#define VOC 32000
#define EMB 512
#define BSZ 8
#define SEQ 1024
#define GATES 2048           // 4*H
#define BT (BSZ * SEQ)       // 8192
#define NB 32                // persistent blocks in recurrent kernel
#define BK 64                // K-stage for LDS-staged GEMM
#define LDA 72               // padded LDS row stride (elements, 144B = 16B-aligned)

typedef __attribute__((ext_vector_type(16))) __bf16 v16bf;
typedef __attribute__((ext_vector_type(8)))  float  v8f;

union Frag16 {
    uint4 u[2];   // 32 bytes = 16 bf16
    v16bf v;
};

__device__ __forceinline__ uint16_t f32_to_bf16_rne(float x) {
    uint32_t b = __float_as_uint(x);
    uint32_t r = b + 0x7FFFu + ((b >> 16) & 1u);
    return (uint16_t)(r >> 16);
}

__device__ __forceinline__ float sigmoidf_(float x) {
    return 1.0f / (1.0f + __expf(-x));
}

// ---------------------------------------------------------------------------
// CDNA5 async Global -> LDS copy (per-lane 16B), tracked by ASYNCcnt.
// LDS operand: low 32 bits of the generic shared pointer == LDS byte offset
// (flat shared addresses are SHARED_BASE<<32 | lds_offset per ISA aperture).
// ---------------------------------------------------------------------------
__device__ __forceinline__ void async_copy_b128(const uint16_t* g, uint16_t* lds) {
    uint32_t laddr = (uint32_t)(uintptr_t)lds;
    uint64_t gaddr = (uint64_t)(uintptr_t)g;
    asm volatile("global_load_async_to_lds_b128 %0, %1, off"
                 :: "v"(laddr), "v"(gaddr) : "memory");
}

__device__ __forceinline__ void async_wait0() {
#if __has_builtin(__builtin_amdgcn_s_wait_asynccnt)
    __builtin_amdgcn_s_wait_asynccnt(0);
#else
    asm volatile("s_wait_asynccnt 0x0" ::: "memory");
#endif
}

// Stage a 128 x 64 bf16 tile (rows row0..row0+127, cols k0..k0+63 of a
// row-major [*,512] matrix) into LDS with padded stride LDA.
// 1024 16B chunks / 256 threads = 4 async loads per thread.
__device__ __forceinline__ void stage_tile_async(const uint16_t* __restrict__ g,
                                                 int row0, int k0,
                                                 uint16_t* lds, int tid) {
#pragma unroll
    for (int i = 0; i < 4; ++i) {
        int c   = tid + i * 256;
        int row = c >> 3;
        int ch  = (c & 7) << 3;    // element offset within the 64-col slab
        async_copy_b128(g + (size_t)(row0 + row) * EMB + k0 + ch,
                        lds + row * LDA + ch);
    }
}

// ---------------------------------------------------------------------------
// f32 -> bf16 bulk convert (weights), grid-stride
// ---------------------------------------------------------------------------
__global__ void k_f32_to_bf16(const float* __restrict__ src,
                              uint16_t* __restrict__ dst, long n) {
    long stride = (long)gridDim.x * blockDim.x;
    for (long i = (long)blockIdx.x * blockDim.x + threadIdx.x; i < n; i += stride)
        dst[i] = f32_to_bf16_rne(src[i]);
}

// ---------------------------------------------------------------------------
// Embedding gather + convert: x0[bt][e] = bf16(wte[inp[bt]][e])
// ---------------------------------------------------------------------------
__global__ void k_embed(const int* __restrict__ inp, const float* __restrict__ wte,
                        uint16_t* __restrict__ x0) {
    int bt = blockIdx.x;                       // 0..8191
    int tok = inp[bt];
    const float* src = wte + (size_t)tok * EMB;
    uint16_t* dst = x0 + (size_t)bt * EMB;
    for (int e = threadIdx.x; e < EMB; e += blockDim.x)
        dst[e] = f32_to_bf16_rne(src[e]);
}

// ---------------------------------------------------------------------------
// Fragment loader from an LDS tile (padded stride LDA), per CDNA5 ISA
// 16-bit A 16x32 layout: lane l -> row l&15 ; K = {kb..kb+7, kb+16..kb+23},
// kb = (l<16 ? 0 : 8). B mirrors A with lane -> N, so weight rows use the
// same loader (GEMMs are x * W^T with W row-major [N,512]).
// ---------------------------------------------------------------------------
__device__ __forceinline__ void load_frag_lds(Frag16& f, const uint16_t* t,
                                              int row, int k0, int lane) {
    int r  = row + (lane & 15);
    int kb = k0 + ((lane >> 4) << 3);
    const uint16_t* base = t + r * LDA + kb;   // 16B aligned (LDA*2 = 144)
    f.u[0] = *(const uint4*)(base);
    f.u[1] = *(const uint4*)(base + 16);
}

// ---------------------------------------------------------------------------
// LDS-staged, double-buffered WMMA GEMM:
//   C[M,N] = A[M,512] * W[N,512]^T (+bias0[n]+bias1[n])
// block = 256 threads = 8 waves; block tile 128x128; wave tile 64x32;
// K staged in 64-wide double-buffered LDS slabs filled with async copies.
// ---------------------------------------------------------------------------
__global__ __launch_bounds__(256)
void k_gemm_bf16(const uint16_t* __restrict__ A,
                 const uint16_t* __restrict__ W,
                 const float* __restrict__ bias0,
                 const float* __restrict__ bias1,
                 float* __restrict__ C, int M, int N) {
    __shared__ uint16_t Asm[2][128 * LDA];
    __shared__ uint16_t Bsm[2][128 * LDA];

    const int tid  = threadIdx.x;
    const int lane = tid & 31;
    const int wave = tid >> 5;
    const int wm = wave >> 2;                  // 0..1
    const int wn = wave & 3;                   // 0..3
    const int blkM = blockIdx.y * 128;
    const int blkN = blockIdx.x * 128;

    v8f acc[4][2];
#pragma unroll
    for (int i = 0; i < 4; ++i)
#pragma unroll
        for (int j = 0; j < 2; ++j) acc[i][j] = (v8f)(0.0f);

    // prologue: stage K-slab 0 into buffer 0
    stage_tile_async(A, blkM, 0, &Asm[0][0], tid);
    stage_tile_async(W, blkN, 0, &Bsm[0][0], tid);
    async_wait0();
    __syncthreads();

    for (int ks = 0; ks < EMB / BK; ++ks) {            // 8 stages
        const int cur = ks & 1;
        if (ks + 1 < EMB / BK) {                       // overlap next stage
            stage_tile_async(A, blkM, (ks + 1) * BK, &Asm[cur ^ 1][0], tid);
            stage_tile_async(W, blkN, (ks + 1) * BK, &Bsm[cur ^ 1][0], tid);
        }
#pragma unroll
        for (int sub = 0; sub < 2; ++sub) {            // two 32-K WMMA steps
            const int k0 = sub * 32;
            Frag16 a[4], b[2];
#pragma unroll
            for (int i = 0; i < 4; ++i)
                load_frag_lds(a[i], &Asm[cur][0], wm * 64 + 16 * i, k0, lane);
#pragma unroll
            for (int j = 0; j < 2; ++j)
                load_frag_lds(b[j], &Bsm[cur][0], wn * 32 + 16 * j, k0, lane);
#pragma unroll
            for (int i = 0; i < 4; ++i)
#pragma unroll
                for (int j = 0; j < 2; ++j)
                    acc[i][j] = __builtin_amdgcn_wmma_f32_16x16x32_bf16(
                        false, a[i].v, false, b[j].v, (short)0, acc[i][j], false, false);
        }
        async_wait0();
        __syncthreads();
    }

    // D layout: M = v + 8*(lane>=16), N = lane&15
    const int nlo  = lane & 15;
    const int rsel = (lane >> 4) * 8;
#pragma unroll
    for (int j = 0; j < 2; ++j) {
        int n = blkN + wn * 32 + 16 * j + nlo;
        float bias = 0.0f;
        if (bias0) bias += bias0[n];
        if (bias1) bias += bias1[n];
#pragma unroll
        for (int i = 0; i < 4; ++i) {
#pragma unroll
            for (int v = 0; v < 8; ++v) {
                int m = blkM + wm * 64 + 16 * i + rsel + v;
                C[(size_t)m * N + n] = acc[i][j][v] + bias;
            }
        }
    }
}

// ---------------------------------------------------------------------------
// Zero h-state and per-step barrier counters before each recurrent pass
// ---------------------------------------------------------------------------
__global__ void k_init(uint16_t* __restrict__ hstate, int* __restrict__ bars) {
    int i = blockIdx.x * blockDim.x + threadIdx.x;
    if (i < BSZ * EMB) hstate[i] = 0;
    if (i < SEQ) bars[i] = 0;
}

// ---------------------------------------------------------------------------
// Persistent LSTM recurrence for one layer.
// NB=32 blocks, 128 threads (4 waves). Block b owns H columns [16b, 16b+16).
// W_hh rows for its 4 gate slices (64 rows x 512 bf16 = 64KB) stay in LDS.
// Per step: stage h_{t-1} (8KB) in LDS, each wave computes one gate tile
// (M=16 pads batch 8, N=16, K=512 -> 16 chained WMMAs), exchange gates via
// LDS, elementwise cell update (c kept in LDS), publish h chunk as bf16,
// grid-wide arrive+spin barrier on bars[t].
// ---------------------------------------------------------------------------
__global__ __launch_bounds__(128)
void k_lstm_rec(const float* __restrict__ xg,        // [8,1024,2048]
                const uint16_t* __restrict__ whh,    // [2048,512] bf16 (layer)
                uint16_t* __restrict__ hstate,       // [8,512] bf16
                uint16_t* __restrict__ xout,         // [8,1024,512] bf16
                int* __restrict__ bars) {            // [1024]
    __shared__ uint16_t Bsh[64 * EMB];               // 64KB weights
    __shared__ uint16_t Ash[16 * EMB];               // 16KB h (rows 8..15 zero)
    __shared__ float    g_lds[4][BSZ][16];
    __shared__ float    c_lds[BSZ][16];

    const int tid  = threadIdx.x;
    const int lane = tid & 31;
    const int q    = tid >> 5;                       // wave id == gate (i,f,g,o)
    const int hc0  = blockIdx.x * 16;

    // stage W_hh rows: LDS row (q'*16+j) <- global row (q'*512 + hc0 + j)
    for (int c = tid; c < 64 * 64; c += blockDim.x) {
        int lrow = c >> 6;
        int koff = (c & 63) * 8;
        int grow = (lrow >> 4) * EMB + hc0 + (lrow & 15);
        *(uint4*)&Bsh[lrow * EMB + koff] = *(const uint4*)&whh[(size_t)grow * EMB + koff];
    }
    for (int c = tid; c < 8 * EMB; c += blockDim.x) Ash[8 * EMB + c] = 0;  // pad rows
    if (tid < BSZ * 16) c_lds[tid >> 4][tid & 15] = 0.0f;
    __syncthreads();

    for (int t = 0; t < SEQ; ++t) {
        // stage h_{t-1} (8 rows x 512 bf16)
        for (int c = tid; c < 8 * 64; c += blockDim.x) {
            int row = c >> 6, koff = (c & 63) * 8;
            *(uint4*)&Ash[row * EMB + koff] = *(const uint4*)&hstate[row * EMB + koff];
        }
        __syncthreads();

        v8f acc = (v8f)(0.0f);
#pragma unroll 4
        for (int kt = 0; kt < 16; ++kt) {
            int k0 = kt * 32;
            int r  = lane & 15;
            int kb = k0 + ((lane >> 4) << 3);
            Frag16 a, b;
            const uint16_t* ap = &Ash[r * EMB + kb];
            a.u[0] = *(const uint4*)ap;
            a.u[1] = *(const uint4*)(ap + 16);
            const uint16_t* bp = &Bsh[(q * 16 + r) * EMB + kb];
            b.u[0] = *(const uint4*)bp;
            b.u[1] = *(const uint4*)(bp + 16);
            acc = __builtin_amdgcn_wmma_f32_16x16x32_bf16(
                false, a.v, false, b.v, (short)0, acc, false, false);
        }

        // valid rows are lanes 0..15 (M = v = batch); add xg, park in LDS
        if (lane < 16) {
            int gc = q * EMB + hc0 + lane;     // gate column in [0,2048)
#pragma unroll
            for (int v = 0; v < 8; ++v)
                g_lds[q][v][lane] = acc[v] + xg[((size_t)v * SEQ + t) * GATES + gc];
        }
        __syncthreads();

        // elementwise cell update: one thread per (batch, col)
        {
            int bb = tid >> 4, cc = tid & 15;
            float i_g = g_lds[0][bb][cc];
            float f_g = g_lds[1][bb][cc];
            float g_g = g_lds[2][bb][cc];
            float o_g = g_lds[3][bb][cc];
            float cn  = sigmoidf_(f_g) * c_lds[bb][cc] + sigmoidf_(i_g) * tanhf(g_g);
            float h   = sigmoidf_(o_g) * tanhf(cn);
            c_lds[bb][cc] = cn;
            uint16_t hb = f32_to_bf16_rne(h);
            hstate[bb * EMB + hc0 + cc] = hb;
            xout[((size_t)bb * SEQ + t) * EMB + hc0 + cc] = hb;
        }
        __syncthreads();
        __threadfence();
        if (tid == 0) {
            __hip_atomic_fetch_add(&bars[t], 1, __ATOMIC_RELEASE, __HIP_MEMORY_SCOPE_AGENT);
            while (__hip_atomic_load(&bars[t], __ATOMIC_ACQUIRE, __HIP_MEMORY_SCOPE_AGENT) < NB)
                __builtin_amdgcn_s_sleep(1);
        }
        __syncthreads();
    }
}

// ---------------------------------------------------------------------------
// Launch: convert -> embed -> [xg GEMM -> recur] x2 -> FC GEMM
// ---------------------------------------------------------------------------
extern "C" void kernel_launch(void* const* d_in, const int* in_sizes, int n_in,
                              void* d_out, int out_size, void* d_ws, size_t ws_size,
                              hipStream_t stream) {
    const int*   inp  = (const int*)d_in[0];
    const float* wte  = (const float*)d_in[1];
    const float* W_ih = (const float*)d_in[2];   // [2,2048,512]
    const float* W_hh = (const float*)d_in[3];   // [2,2048,512]
    const float* b_ih = (const float*)d_in[4];   // [2,2048]
    const float* b_hh = (const float*)d_in[5];   // [2,2048]
    const float* fc_w = (const float*)d_in[6];   // [32000,512]
    const float* fc_b = (const float*)d_in[7];   // [32000]
    float* out = (float*)d_out;                  // [8,1024,32000]

    char* ws = (char*)d_ws;
    uint16_t* wih_bf = (uint16_t*)ws;  ws += (size_t)2 * GATES * EMB * 2;
    uint16_t* whh_bf = (uint16_t*)ws;  ws += (size_t)2 * GATES * EMB * 2;
    uint16_t* fc_bf  = (uint16_t*)ws;  ws += (size_t)VOC * EMB * 2;
    uint16_t* x0     = (uint16_t*)ws;  ws += (size_t)BT * EMB * 2;
    uint16_t* x1     = (uint16_t*)ws;  ws += (size_t)BT * EMB * 2;
    uint16_t* x2     = (uint16_t*)ws;  ws += (size_t)BT * EMB * 2;
    float*    xg     = (float*)ws;     ws += (size_t)BT * GATES * 4;
    uint16_t* hstate = (uint16_t*)ws;  ws += (size_t)BSZ * EMB * 2;
    int*      bars   = (int*)ws;       ws += (size_t)SEQ * 4;

    // weight conversions to bf16
    k_f32_to_bf16<<<2048, 256, 0, stream>>>(W_ih, wih_bf, (long)2 * GATES * EMB);
    k_f32_to_bf16<<<2048, 256, 0, stream>>>(W_hh, whh_bf, (long)2 * GATES * EMB);
    k_f32_to_bf16<<<8192, 256, 0, stream>>>(fc_w, fc_bf, (long)VOC * EMB);
    k_embed<<<BT, 128, 0, stream>>>(inp, wte, x0);

    dim3 blk(256);
    dim3 g_xg(GATES / 128, BT / 128);            // (16, 64)
    dim3 g_fc(VOC / 128, BT / 128);              // (250, 64)

    // layer 0
    k_gemm_bf16<<<g_xg, blk, 0, stream>>>(x0, wih_bf, b_ih, b_hh, xg, BT, GATES);
    k_init<<<(BSZ * EMB + 255) / 256, 256, 0, stream>>>(hstate, bars);
    k_lstm_rec<<<NB, 128, 0, stream>>>(xg, whh_bf, hstate, x1, bars);

    // layer 1
    k_gemm_bf16<<<g_xg, blk, 0, stream>>>(x1, wih_bf + (size_t)GATES * EMB,
                                          b_ih + GATES, b_hh + GATES, xg, BT, GATES);
    k_init<<<(BSZ * EMB + 255) / 256, 256, 0, stream>>>(hstate, bars);
    k_lstm_rec<<<NB, 128, 0, stream>>>(xg, whh_bf + (size_t)GATES * EMB, hstate, x2, bars);

    // vocab projection (dominant GEMM: 268 GFLOP, 1.05 GB output)
    k_gemm_bf16<<<g_fc, blk, 0, stream>>>(x2, fc_bf, fc_b, nullptr, out, BT, VOC);
}
